// SimplePerformerSeqModel_25331717112442
// MI455X (gfx1250) — matrix operand order repaired and checked by reading, not compiled, not run on hardware
//
#include <hip/hip_runtime.h>
#include <hip/hip_bf16.h>

// ---------------------------------------------------------------------------
// SimplePerformerSeqModel forward for MI455X (gfx1250, wave32, WMMA).
// All GEMMs: bf16 inputs (converted during LDS staging), fp32 accumulate via
// v_wmma_f32_16x16x32_bf16.  Elementwise ops fused into GEMM epilogues.
// ---------------------------------------------------------------------------

#define B_  2
#define S_  4096
#define D_  256
#define P_  64
#define L_  2
#define V_  32000
#define F_  1024
#define BS_ (B_ * S_)

typedef __bf16 v16bf __attribute__((ext_vector_type(16)));
typedef float  v8f   __attribute__((ext_vector_type(8)));
typedef float  f32x4 __attribute__((ext_vector_type(4)));
typedef unsigned int u32x4 __attribute__((ext_vector_type(4)));

enum : int { EPI_NONE = 0, EPI_ELU1 = 1, EPI_BIAS_GELU = 2, EPI_BIAS_RES = 3, EPI_ATOMIC = 4 };

#define BM 128
#define BN 64
#define BK 32
#define LDA_S 40   // padded A-tile row stride (bf16 elems): 80B, rows stay 16B-aligned

// C[M,N] += A[M,K] (fp32, optional transposed source) * B[K,N] (fp32),
// bf16 WMMA with fp32 accumulation.  Block: 256 threads = 8 wave32.
// Wave w computes rows [w*16, w*16+16) of the 128x64 block tile (4 WMMAs wide).
// Constraints used by the launcher: K, ksplit multiples of 32; N multiple of 64;
// M multiple of 16; !TRANSA implies M multiple of 128 (no A-row guards needed).
template <int EPI, bool TRANSA>
__global__ __launch_bounds__(256) void gemm_kernel(
    const float* __restrict__ A, int lda,
    const float* __restrict__ Bm, int ldb,
    float* __restrict__ C, int ldc,
    const float* __restrict__ bias,
    const float* __restrict__ resid,
    int M, int N, int K, int ksplit)
{
  __shared__ __attribute__((aligned(32))) __bf16 Alds[BM * LDA_S];
  __shared__ __attribute__((aligned(32))) __bf16 Bswz[4 * 32 * 16];  // fragment-ready

  const int t    = threadIdx.x;
  const int lane = t & 31;
  const int warp = t >> 5;        // 0..7
  const int half = lane >> 4;     // 0/1 (lane half selects K sub-range per ISA layout)
  const int r    = lane & 15;

  const int m0 = blockIdx.y * BM;
  const int n0 = blockIdx.x * BN;
  const int kbeg = blockIdx.z * ksplit;
  int kend = kbeg + ksplit; if (kend > K) kend = K;

  v8f acc[4] = {};

  // staging geometry (loop-invariant)
  const int arow = t >> 1;              // 0..127 (row in A tile, !TRANSA)
  const int ac0  = (t & 1) * 16;        // col group (!TRANSA)
  const int tam  = t & (BM - 1);        // m within tile (TRANSA); invariant across i
  const bool tain = TRANSA && ((m0 + tam) < M);
  // B staging: each thread owns one column n and 8 consecutive k values, so the
  // swizzled-fragment destination is 16 contiguous bytes -> one ds_store_b128.
  // Frag layout (32x16, 16-bit): lane = ((k>>3)&1)*16 + n%16 holds col n, elem j=(k&7)+(k>>4)*8.
  const int bn  = t & 63;               // column within the 64-wide tile
  const int bkg = (t >> 6) * 8;         // k group: 0,8,16,24
  __bf16* bdst = &Bswz[((((bn >> 4) * 32) + (((bkg >> 3) & 1) << 4) + (bn & 15)) << 4) + ((bkg >> 4) << 3)];

  for (int k0 = kbeg; k0 < kend; k0 += BK) {
    // ---- stage A tile: 128x32 fp32 -> bf16 ----
    if (!TRANSA) {
      const float* ap = A + (size_t)(m0 + arow) * lda + k0 + ac0;
      f32x4 f0 = *(const f32x4*)(ap + 0);
      f32x4 f1 = *(const f32x4*)(ap + 4);
      f32x4 f2 = *(const f32x4*)(ap + 8);
      f32x4 f3 = *(const f32x4*)(ap + 12);
      __bf16 tmp[16];
#pragma unroll
      for (int j = 0; j < 4; ++j) {
        tmp[j]      = (__bf16)f0[j];
        tmp[4 + j]  = (__bf16)f1[j];
        tmp[8 + j]  = (__bf16)f2[j];
        tmp[12 + j] = (__bf16)f3[j];
      }
      __bf16* dst = &Alds[arow * LDA_S + ac0];
      ((u32x4*)dst)[0] = ((const u32x4*)tmp)[0];
      *(u32x4*)(dst + 8) = ((const u32x4*)tmp)[1];
    } else {
      // A source viewed transposed: Alds[m][kk] = A[(k0+kk)*lda + m0+m].
      // Predicate is loop-invariant per thread: one divergent region, not 16.
      if (tain) {
#pragma unroll
        for (int i = 0; i < 16; ++i) {
          int ak = 2 * i + (t >> 7);     // (i*256+t)>>7
          Alds[tam * LDA_S + ak] = (__bf16)A[(size_t)(k0 + ak) * lda + m0 + tam];
        }
      } else {
#pragma unroll
        for (int i = 0; i < 16; ++i) {
          int ak = 2 * i + (t >> 7);
          Alds[tam * LDA_S + ak] = (__bf16)0.0f;
        }
      }
    }
    // ---- stage B tile: 32x64 fp32 -> bf16, swizzled to WMMA-B fragment order.
    // 8 column-strided b32 loads (wave-coalesced along n), one b128 LDS store.
    {
      const float* bp = Bm + (size_t)(k0 + bkg) * ldb + n0 + bn;
      __bf16 tb[8];
#pragma unroll
      for (int j = 0; j < 8; ++j) tb[j] = (__bf16)bp[(size_t)j * ldb];
      *(u32x4*)bdst = *(const u32x4*)tb;
      if (k0 + BK < kend) {  // prefetch next K tile -> global_prefetch_b8
        __builtin_prefetch(bp + (size_t)BK * ldb, 0, 1);
        __builtin_prefetch(TRANSA ? &A[(size_t)(k0 + BK + (t >> 3)) * lda + m0 + tam]
                                  : &A[(size_t)(m0 + arow) * lda + k0 + BK + ac0], 0, 1);
      }
    }
    __syncthreads();

    // ---- A fragment (16x32 bf16): lane half picks K {0..7,16..23} / {8..15,24..31}
    v16bf af;
    {
      const __bf16* ar = &Alds[(warp * 16 + r) * LDA_S + half * 8];
      ((u32x4*)&af)[0] = *(const u32x4*)(ar);
      ((u32x4*)&af)[1] = *(const u32x4*)(ar + 16);
    }
    // gather all 4 B fragments first so waits don't serialize the WMMAs
    v16bf bfrag[4];
#pragma unroll
    for (int c = 0; c < 4; ++c) {
      const __bf16* br = &Bswz[(c * 32 + lane) * 16];
      ((u32x4*)&bfrag[c])[0] = ((const u32x4*)br)[0];
      ((u32x4*)&bfrag[c])[1] = ((const u32x4*)br)[1];
    }
#pragma unroll
    for (int c = 0; c < 4; ++c)
      acc[c] = __builtin_amdgcn_wmma_f32_16x16x32_bf16(
          false, af, false, bfrag[c], (short)0, acc[c], false, false);
    __syncthreads();
  }

  // ---- epilogue: C/D layout: VGPR rr -> M = rr + 8*half, N = lane&15 ----
  // M is always a multiple of 16 here, so the guard is wave-uniform.
  if (m0 + warp * 16 < M) {
#pragma unroll
    for (int c = 0; c < 4; ++c) {
#pragma unroll
      for (int rr = 0; rr < 8; ++rr) {
        int row = m0 + warp * 16 + half * 8 + rr;
        int col = n0 + c * 16 + r;
        float x = acc[c][rr];
        if (EPI == EPI_BIAS_GELU || EPI == EPI_BIAS_RES) x += bias[col];
        if (EPI == EPI_ELU1) x = (x > 0.0f) ? (x + 1.0f) : expf(x);  // elu(x)+1
        if (EPI == EPI_BIAS_GELU)
          x = 0.5f * x * (1.0f + erff(x * 0.70710678118654752f));    // exact gelu
        if (EPI == EPI_BIAS_RES) x += resid[(size_t)row * ldc + col];
        if (EPI == EPI_ATOMIC) atomicAdd(&C[(size_t)row * ldc + col], x);
        else                   C[(size_t)row * ldc + col] = x;
      }
    }
  }
}

// h[b,s,:] = embed[x[b,s],:] + pos[s,:]
__global__ __launch_bounds__(256) void embed_kernel(
    const int* __restrict__ x, const float* __restrict__ embed,
    const float* __restrict__ pos, float* __restrict__ h)
{
  int row = blockIdx.x;          // 0..BS-1
  int d   = threadIdx.x;         // 0..255
  int s   = row & (S_ - 1);
  int tok = x[row];
  h[(size_t)row * D_ + d] = embed[(size_t)tok * D_ + d] + pos[(size_t)s * D_ + d];
}

__global__ __launch_bounds__(256) void zero_kernel(float* __restrict__ p, int n) {
  int i = blockIdx.x * 256 + threadIdx.x;
  if (i < n) p[i] = 0.0f;
}

// a = num[row,:] * 1/(sum(qp[row,:])+1e-8), then LayerNorm(a)*g+b -> out
__global__ __launch_bounds__(256) void attn_ln_kernel(
    const float* __restrict__ num, const float* __restrict__ qp,
    const float* __restrict__ g, const float* __restrict__ b,
    float* __restrict__ out)
{
  __shared__ float sb[256];
  int row = blockIdx.x;
  int t   = threadIdx.x;

  sb[t] = (t < P_) ? qp[(size_t)row * P_ + t] : 0.0f;
  __syncthreads();
  for (int s = 128; s > 0; s >>= 1) { if (t < s) sb[t] += sb[t + s]; __syncthreads(); }
  float z = 1.0f / (sb[0] + 1e-8f);
  __syncthreads();

  float a = num[(size_t)row * D_ + t] * z;
  sb[t] = a;
  __syncthreads();
  for (int s = 128; s > 0; s >>= 1) { if (t < s) sb[t] += sb[t + s]; __syncthreads(); }
  float mu = sb[0] * (1.0f / D_);
  __syncthreads();

  float d = a - mu;
  sb[t] = d * d;
  __syncthreads();
  for (int s = 128; s > 0; s >>= 1) { if (t < s) sb[t] += sb[t + s]; __syncthreads(); }
  float var = sb[0] * (1.0f / D_);

  out[(size_t)row * D_ + t] = d * rsqrtf(var + 1e-5f) * g[t] + b[t];
}

// ---------------------------------------------------------------------------

template <int EPI, bool TRANSA>
static void launch_gemm(hipStream_t st,
                        const float* A, int lda, const float* Bm, int ldb,
                        float* C, int ldc, const float* bias, const float* resid,
                        int M, int N, int K, int ksplit, int nz)
{
  dim3 grid(N / BN, (M + BM - 1) / BM, nz);
  gemm_kernel<EPI, TRANSA><<<grid, dim3(256), 0, st>>>(
      A, lda, Bm, ldb, C, ldc, bias, resid, M, N, K, ksplit);
}

extern "C" void kernel_launch(void* const* d_in, const int* in_sizes, int n_in,
                              void* d_out, int out_size, void* d_ws, size_t ws_size,
                              hipStream_t stream)
{
  (void)in_sizes; (void)n_in; (void)out_size; (void)ws_size;

  const int*   x     = (const int*)  d_in[0];
  const float* embed = (const float*)d_in[1];
  const float* pos   = (const float*)d_in[2];
  const float* Wq    = (const float*)d_in[3];
  const float* Wk    = (const float*)d_in[4];
  const float* Wv    = (const float*)d_in[5];
  const float* proj  = (const float*)d_in[6];
  const float* ln_g  = (const float*)d_in[7];
  const float* ln_b  = (const float*)d_in[8];
  const float* W1    = (const float*)d_in[9];
  const float* b1    = (const float*)d_in[10];
  const float* W2    = (const float*)d_in[11];
  const float* b2    = (const float*)d_in[12];
  const float* Wout  = (const float*)d_in[13];
  float*       out   = (float*)d_out;

  // workspace layout (fp32 elements); f1 aliases q..num (32MB region)
  float* ws   = (float*)d_ws;
  float* h    = ws;                          // BS*D
  float* q    = h    + (size_t)BS_ * D_;     // BS*D
  float* k    = q    + (size_t)BS_ * D_;     // BS*D
  float* v    = k    + (size_t)BS_ * D_;     // BS*D
  float* numb = v    + (size_t)BS_ * D_;     // BS*D
  float* f1   = q;                           // BS*F (aliases q,k,v,num)
  float* qp   = numb + (size_t)BS_ * D_;     // BS*P
  float* kp   = qp   + (size_t)BS_ * P_;     // BS*P
  float* kv   = kp   + (size_t)BS_ * P_;     // B*P*D
  float* a_ln = kv   + (size_t)B_ * P_ * D_; // BS*D

  // h = embed[x] + pos
  embed_kernel<<<dim3(BS_), dim3(256), 0, stream>>>(x, embed, pos, h);

  for (int i = 0; i < L_; ++i) {
    const float* Wqi = Wq + (size_t)i * D_ * D_;
    const float* Wki = Wk + (size_t)i * D_ * D_;
    const float* Wvi = Wv + (size_t)i * D_ * D_;
    const float* pji = proj + (size_t)i * D_ * P_;
    const float* W1i = W1 + (size_t)i * D_ * F_;
    const float* b1i = b1 + (size_t)i * F_;
    const float* W2i = W2 + (size_t)i * F_ * D_;
    const float* b2i = b2 + (size_t)i * D_;

    // q,k,v = h @ W{q,k,v}        [8192,256] x [256,256]
    launch_gemm<EPI_NONE, false>(stream, h, D_, Wqi, D_, q, D_, nullptr, nullptr, BS_, D_, D_, D_, 1);
    launch_gemm<EPI_NONE, false>(stream, h, D_, Wki, D_, k, D_, nullptr, nullptr, BS_, D_, D_, D_, 1);
    launch_gemm<EPI_NONE, false>(stream, h, D_, Wvi, D_, v, D_, nullptr, nullptr, BS_, D_, D_, D_, 1);

    // q' = elu(q@proj)+1 ; k' = elu(k@proj)+1     [8192,256] x [256,64]
    launch_gemm<EPI_ELU1, false>(stream, q, D_, pji, P_, qp, P_, nullptr, nullptr, BS_, P_, D_, D_, 1);
    launch_gemm<EPI_ELU1, false>(stream, k, D_, pji, P_, kp, P_, nullptr, nullptr, BS_, P_, D_, D_, 1);

    // kv[b] = k'[b]^T @ v[b]      [64,4096] x [4096,256], split-K with f32 atomics
    zero_kernel<<<dim3((B_ * P_ * D_ + 255) / 256), dim3(256), 0, stream>>>(kv, B_ * P_ * D_);
    for (int bb = 0; bb < B_; ++bb)
      launch_gemm<EPI_ATOMIC, true>(stream,
          kp + (size_t)bb * S_ * P_, P_, v + (size_t)bb * S_ * D_, D_,
          kv + (size_t)bb * P_ * D_, D_, nullptr, nullptr, P_, D_, S_, S_ / 8, 8);

    // num[b] = q'[b] @ kv[b]      [4096,64] x [64,256]
    for (int bb = 0; bb < B_; ++bb)
      launch_gemm<EPI_NONE, false>(stream,
          qp + (size_t)bb * S_ * P_, P_, kv + (size_t)bb * P_ * D_, D_,
          numb + (size_t)bb * S_ * D_, D_, nullptr, nullptr, S_, D_, P_, P_, 1);

    // a = LN(num * z) * g + b
    attn_ln_kernel<<<dim3(BS_), dim3(256), 0, stream>>>(
        numb, qp, ln_g + (size_t)i * D_, ln_b + (size_t)i * D_, a_ln);

    // f1 = gelu(a @ W1 + b1)      [8192,256] x [256,1024]
    launch_gemm<EPI_BIAS_GELU, false>(stream, a_ln, D_, W1i, F_, f1, F_, b1i, nullptr, BS_, F_, D_, D_, 1);

    // h = f1 @ W2 + b2 + h        [8192,1024] x [1024,256]
    launch_gemm<EPI_BIAS_RES, false>(stream, f1, F_, W2i, D_, h, D_, b2i, h, BS_, D_, F_, F_, 1);
  }

  // logits = h @ Wout             [8192,256] x [256,32000]
  launch_gemm<EPI_NONE, false>(stream, h, D_, Wout, V_, out, V_, nullptr, nullptr, BS_, V_, D_, D_, 1);
}
